// GraphDiffusionBackbone_69887707840999
// MI455X (gfx1250) — compile-verified, matrix operand order checked by reading, not run on hardware
//
#include <hip/hip_runtime.h>
#include <hip/hip_fp16.h>

// ---------------- CDNA5 WMMA types ----------------
typedef __attribute__((ext_vector_type(16))) _Float16 v16h;
typedef __attribute__((ext_vector_type(8)))  _Float16 v8h;
typedef __attribute__((ext_vector_type(8)))  float    v8f;

#define BQ 8
#define NQ 128
#define HQ 768
#define NHQ 12
#define DHQ 64
#define FFQ 3072
#define EVQ 8

__device__ __forceinline__ float gelu_f(float x) {
    return 0.5f * x * (1.0f + erff(x * 0.70710678118654752440f));
}

// Per-lane fragment load for the CDNA5 16-bit 16x32 operand layout:
// lanes 0-15 hold K {0..7} and {16..23}; lanes 16-31 hold K {8..15} and {24..31}.
// Both chunks are 16-byte contiguous loads from a row-major row pointer.
__device__ __forceinline__ v16h frag_ld(const _Float16* __restrict__ row, int k0, int hs) {
    v8h lo = *(const v8h*)(row + k0 + hs);
    v8h hi = *(const v8h*)(row + k0 + 16 + hs);
    return __builtin_shufflevector(lo, hi, 0, 1, 2, 3, 4, 5, 6, 7,
                                           8, 9, 10, 11, 12, 13, 14, 15);
}

// ---------------- embedding: x = node_emb[X] + time_emb[t] ----------------
__global__ void embed_kernel(const int* __restrict__ X, const int* __restrict__ t,
                             const float* __restrict__ node_emb,
                             const float* __restrict__ time_emb,
                             float* __restrict__ x) {
    int idx = blockIdx.x * 256 + threadIdx.x;
    if (idx >= BQ * NQ * HQ) return;
    int h = idx % HQ;
    int bn = idx / HQ;
    int b = bn / NQ;
    x[idx] = node_emb[X[bn] * HQ + h] + time_emb[t[b] * HQ + h];
}

// ---------------- LayerNorm (f32 in, f16 out) ----------------
__global__ void ln_f16_kernel(const float* __restrict__ x, const float* __restrict__ g,
                              const float* __restrict__ b, _Float16* __restrict__ out) {
    int row = blockIdx.x;
    int tid = threadIdx.x;
    __shared__ float red[256];
    const float* xr = x + row * HQ;
    float s = 0.f, s2 = 0.f;
    for (int h = tid; h < HQ; h += 256) { float v = xr[h]; s += v; s2 += v * v; }
    red[tid] = s; __syncthreads();
    for (int off = 128; off > 0; off >>= 1) { if (tid < off) red[tid] += red[tid + off]; __syncthreads(); }
    float mean = red[0] / HQ;
    __syncthreads();
    red[tid] = s2; __syncthreads();
    for (int off = 128; off > 0; off >>= 1) { if (tid < off) red[tid] += red[tid + off]; __syncthreads(); }
    float var = red[0] / HQ - mean * mean;
    float rstd = rsqrtf(var + 1e-5f);
    for (int h = tid; h < HQ; h += 256)
        out[row * HQ + h] = (_Float16)((xr[h] - mean) * rstd * g[h] + b[h]);
}

// ---------------- transpose + f32->f16 weight conversion ----------------
// in: (R rows, C cols) f32 row-major.  out: (C rows, R cols) f16 row-major.
__global__ void convT_kernel(const float* __restrict__ in, _Float16* __restrict__ out,
                             int R, int C) {
    int idx = blockIdx.x * 256 + threadIdx.x;
    if (idx >= R * C) return;
    int c = idx / R;
    int r = idx % R;
    out[idx] = (_Float16)in[r * C + c];
}

// ---------------- generic WMMA GEMM: C = A(MxK) * Bt(NxK)^T (+bias) ----------------
// Register-blocked: each wave computes a 16 x (16*NT) strip -> one A fragment
// load feeds NT wmma ops on independent accumulators.
// modes: 0 = f16 out (+bias), 1 = f16 out gelu(+bias),
//        2 = f32 out += acc+bias (residual in place), 3 = f32 out = acc(+bias)
template <int NT>
__global__ void wmma_gemm_tn(const _Float16* __restrict__ A, const _Float16* __restrict__ Bt,
                             const float* __restrict__ bias, float* __restrict__ outf,
                             _Float16* __restrict__ outh, int Mdim, int Ndim, int K,
                             int mode) {
    int lane = threadIdx.x & 31;
    int wave = threadIdx.x >> 5;
    int ngroups = Ndim / (16 * NT);
    int group = blockIdx.x * 4 + wave;
    if (group >= (Mdim >> 4) * ngroups) return;   // wave-uniform: EXEC all-ones at wmma
    int mt = group / ngroups, ng = group % ngroups;
    int l15 = lane & 15;
    int hs = (lane & 16) ? 8 : 0;
    const _Float16* arow = A + (size_t)(mt * 16 + l15) * K;
    const _Float16* brow[NT];
#pragma unroll
    for (int tTile = 0; tTile < NT; ++tTile)
        brow[tTile] = Bt + (size_t)((ng * NT + tTile) * 16 + l15) * K;
    v8f c[NT] = {};
    for (int k0 = 0; k0 < K; k0 += 32) {
        __builtin_prefetch(arow + k0 + 64, 0, 1);   // global_prefetch_b8
        v16h a = frag_ld(arow, k0, hs);
#pragma unroll
        for (int tTile = 0; tTile < NT; ++tTile) {
            v16h b = frag_ld(brow[tTile], k0, hs);
            c[tTile] = __builtin_amdgcn_wmma_f32_16x16x32_f16(false, a, false, b, (short)0,
                                                              c[tTile], false, false);
        }
    }
    int rbase = mt * 16 + ((lane & 16) ? 8 : 0);
#pragma unroll
    for (int tTile = 0; tTile < NT; ++tTile) {
        int col = (ng * NT + tTile) * 16 + l15;
        float bvv = bias ? bias[col] : 0.f;
#pragma unroll
        for (int v = 0; v < 8; ++v) {
            size_t idx = (size_t)(rbase + v) * Ndim + col;
            float val = c[tTile][v] + bvv;
            if (mode == 0)      outh[idx] = (_Float16)val;
            else if (mode == 1) outh[idx] = (_Float16)gelu_f(val);
            else if (mode == 2) outf[idx] = outf[idx] + val;
            else                outf[idx] = val;
        }
    }
}

// ---------------- attention: per (b, head, row-quarter) block ----------------
// S = QK^T*scale + edge bias, masks, softmax -> P(f16) -> O = P*V   (all WMMA)
__global__ void attn_kernel(const _Float16* __restrict__ Q, const _Float16* __restrict__ Kk,
                            const _Float16* __restrict__ Vv, const int* __restrict__ E,
                            const float* __restrict__ Mask, const float* __restrict__ ee,
                            _Float16* __restrict__ O) {
    int bid = blockIdx.x;                 // B*NH*4 blocks
    int quarter = bid & 3;
    int head = (bid >> 2) % NHQ;
    int b = bid / (4 * NHQ);
    int tid = threadIdx.x;                // 128 threads = 4 waves
    int lane = tid & 31;
    int wave = tid >> 5;
    int r0 = quarter * 32;

    __shared__ float    S[32 * 128];      // 16 KB scores
    __shared__ _Float16 P[32 * 128];      //  8 KB probabilities (WMMA A operand)
    __shared__ _Float16 Vt[64 * 128];     // 16 KB V^T (WMMA B operand rows = d)

    // stage V^T for this (b, head)
    for (int idx = tid; idx < 128 * 64; idx += 128) {
        int j = idx >> 6, d = idx & 63;
        Vt[d * 128 + j] = Vv[(size_t)(b * NQ + j) * HQ + head * DHQ + d];
    }

    int l15 = lane & 15;
    int hs = (lane & 16) ? 8 : 0;

    // phase 1: wave -> (m-tile, group of 4 n-tiles); A fragments loaded once,
    // reused across 4 K-tiles (8 wmma per wave).
    {
        int mt = wave >> 1;
        int ng = wave & 1;
        const _Float16* arow = Q + (size_t)(b * NQ + r0 + mt * 16 + l15) * HQ + head * DHQ;
        v16h a0 = frag_ld(arow, 0, hs);
        v16h a1 = frag_ld(arow, 32, hs);
        v8f c[4] = {};
#pragma unroll
        for (int tTile = 0; tTile < 4; ++tTile) {
            int nt = ng * 4 + tTile;
            const _Float16* brow = Kk + (size_t)(b * NQ + nt * 16 + l15) * HQ + head * DHQ;
            c[tTile] = __builtin_amdgcn_wmma_f32_16x16x32_f16(false, a0, false,
                           frag_ld(brow, 0, hs), (short)0, c[tTile], false, false);
            c[tTile] = __builtin_amdgcn_wmma_f32_16x16x32_f16(false, a1, false,
                           frag_ld(brow, 32, hs), (short)0, c[tTile], false, false);
        }
        int rb = mt * 16 + ((lane & 16) ? 8 : 0);
#pragma unroll
        for (int tTile = 0; tTile < 4; ++tTile) {
            int col = (ng * 4 + tTile) * 16 + l15;
#pragma unroll
            for (int v = 0; v < 8; ++v) S[(rb + v) * 128 + col] = c[tTile][v] * 0.125f;
        }
    }
    __syncthreads();

    // phase 2: bias + masks + softmax (one thread per row)
    if (tid < 32) {
        int i = r0 + tid;
        float* Sr = S + tid * 128;
        const int* Er = E + (size_t)(b * NQ + i) * NQ;
        float qv = Mask[b * NQ + i];
        float mx = -3.4e38f;
        for (int j = 0; j < 128; ++j) {
            float s = Sr[j] + ee[Er[j] * NHQ + head];
            if (Mask[b * NQ + j] == 0.f) s = -1e9f;
            if (qv == 0.f) s = -1e9f;
            Sr[j] = s;
            mx = fmaxf(mx, s);
        }
        float sum = 0.f;
        for (int j = 0; j < 128; ++j) { float e = __expf(Sr[j] - mx); Sr[j] = e; sum += e; }
        float inv = 1.f / sum;
        for (int j = 0; j < 128; ++j) P[tid * 128 + j] = (_Float16)(Sr[j] * inv);
    }
    __syncthreads();

    // phase 3: wave -> (m-tile, pair of n-tiles); P fragment reused across 2 V tiles.
    {
        int mt = wave >> 1;
        int np = wave & 1;
        const _Float16* arow = P + (mt * 16 + l15) * 128;
        v8f c[2] = {};
        for (int k0 = 0; k0 < 128; k0 += 32) {
            v16h a = frag_ld(arow, k0, hs);
#pragma unroll
            for (int tTile = 0; tTile < 2; ++tTile) {
                const _Float16* brow = Vt + ((np * 2 + tTile) * 16 + l15) * 128;
                c[tTile] = __builtin_amdgcn_wmma_f32_16x16x32_f16(false, a, false,
                               frag_ld(brow, k0, hs), (short)0, c[tTile], false, false);
            }
        }
        int rb = mt * 16 + ((lane & 16) ? 8 : 0);
#pragma unroll
        for (int tTile = 0; tTile < 2; ++tTile) {
            int col = (np * 2 + tTile) * 16 + l15;
#pragma unroll
            for (int v = 0; v < 8; ++v)
                O[(size_t)(b * NQ + r0 + rb + v) * HQ + head * DHQ + col] = (_Float16)c[tTile][v];
        }
    }
}

// ---------------- fused edge head: el_raw[b,i,j,:] = gelu(ai_i + aj_j) @ W2 + b2 ----------------
__global__ void edge_head_kernel(const float* __restrict__ ai, const float* __restrict__ aj,
                                 const float* __restrict__ W2, const float* __restrict__ b2,
                                 float* __restrict__ elraw) {
    int b = blockIdx.x >> 7;
    int i = blockIdx.x & 127;
    int tid = threadIdx.x;                // 128 threads: thread = j
    __shared__ float ai_sh[HQ];           // 3 KB
    __shared__ float ajc[128 * 64];       // 32 KB chunk of aj
    for (int h = tid; h < HQ; h += 128) ai_sh[h] = ai[(size_t)(b * NQ + i) * HQ + h];
    float acc[EVQ];
#pragma unroll
    for (int e = 0; e < EVQ; ++e) acc[e] = 0.f;
    for (int hc = 0; hc < HQ; hc += 64) {
        __syncthreads();
        for (int idx = tid; idx < 128 * 64; idx += 128) {
            int j2 = idx >> 6, hh = idx & 63;
            ajc[idx] = aj[(size_t)(b * NQ + j2) * HQ + hc + hh];
        }
        __syncthreads();
        for (int hh = 0; hh < 64; ++hh) {
            int h = hc + hh;
            float tv = gelu_f(ai_sh[h] + ajc[tid * 64 + hh]);
            const float* wrow = W2 + h * EVQ;
#pragma unroll
            for (int e = 0; e < EVQ; ++e) acc[e] += tv * wrow[e];
        }
    }
    size_t base = (size_t)((b * NQ + i) * NQ + tid) * EVQ;
#pragma unroll
    for (int e = 0; e < EVQ; ++e) elraw[base + e] = acc[e] + b2[e];
}

// ---------------- symmetrize: out = 0.5*(el + el^T) ----------------
__global__ void symm_kernel(const float* __restrict__ elraw, float* __restrict__ out) {
    int idx = blockIdx.x * 256 + threadIdx.x;
    if (idx >= BQ * NQ * NQ * EVQ) return;
    int e = idx & 7;
    int j = (idx >> 3) & 127;
    int i = (idx >> 10) & 127;
    int b = idx >> 17;
    size_t tidx = (size_t)((b * NQ + j) * NQ + i) * EVQ + e;
    out[idx] = 0.5f * (elraw[idx] + elraw[tidx]);
}

extern "C" void kernel_launch(void* const* d_in, const int* in_sizes, int n_in,
                              void* d_out, int out_size, void* d_ws, size_t ws_size,
                              hipStream_t stream) {
    (void)in_sizes; (void)n_in; (void)out_size; (void)ws_size;
    const int*   X        = (const int*)d_in[0];
    const int*   E        = (const int*)d_in[1];
    const int*   t        = (const int*)d_in[2];
    const float* Mm       = (const float*)d_in[3];
    const float* node_emb = (const float*)d_in[4];
    const float* time_emb = (const float*)d_in[5];
    const float* ln1_g    = (const float*)d_in[6];
    const float* ln1_b    = (const float*)d_in[7];
    const float* Wq       = (const float*)d_in[8];
    const float* bq       = (const float*)d_in[9];
    const float* Wk       = (const float*)d_in[10];
    const float* bk       = (const float*)d_in[11];
    const float* Wv       = (const float*)d_in[12];
    const float* bv       = (const float*)d_in[13];
    const float* Wo       = (const float*)d_in[14];
    const float* bo       = (const float*)d_in[15];
    const float* edge_emb = (const float*)d_in[16];
    const float* ln2_g    = (const float*)d_in[17];
    const float* ln2_b    = (const float*)d_in[18];
    const float* Wf1      = (const float*)d_in[19];
    const float* bf1      = (const float*)d_in[20];
    const float* Wf2      = (const float*)d_in[21];
    const float* bf2      = (const float*)d_in[22];
    const float* lnf_g    = (const float*)d_in[23];
    const float* lnf_b    = (const float*)d_in[24];
    const float* node_W   = (const float*)d_in[25];
    const float* eh_W1    = (const float*)d_in[26];
    const float* eh_b1    = (const float*)d_in[27];
    const float* eh_W2    = (const float*)d_in[28];
    const float* eh_b2    = (const float*)d_in[29];
    float* out = (float*)d_out;

    const int BN = BQ * NQ;                         // 1024 rows
    // ---- workspace carve ----
    char* w = (char*)d_ws;
    float*    x     = (float*)w;    w += (size_t)BN * HQ * 4;
    _Float16* h16   = (_Float16*)w; w += (size_t)BN * HQ * 2;
    _Float16* q16   = (_Float16*)w; w += (size_t)BN * HQ * 2;
    _Float16* k16   = (_Float16*)w; w += (size_t)BN * HQ * 2;
    _Float16* v16   = (_Float16*)w; w += (size_t)BN * HQ * 2;
    _Float16* ao16  = (_Float16*)w; w += (size_t)BN * HQ * 2;
    _Float16* xln16 = (_Float16*)w; w += (size_t)BN * HQ * 2;
    _Float16* ff16  = (_Float16*)w; w += (size_t)BN * FFQ * 2;
    _Float16* wt    = (_Float16*)w; w += (size_t)FFQ * HQ * 2;
    float*    ai    = (float*)w;    w += (size_t)BN * HQ * 4;
    float*    aj    = (float*)w;    w += (size_t)BN * HQ * 4;
    float*    elraw = (float*)w;    w += (size_t)BQ * NQ * NQ * EVQ * 4;

    auto convT = [&](const float* in, _Float16* o, int R, int C) {
        int tot = R * C;
        convT_kernel<<<(tot + 255) / 256, 256, 0, stream>>>(in, o, R, C);
    };
    auto gemm4 = [&](const _Float16* A, const _Float16* Bt, const float* bias,
                     float* of, _Float16* oh, int Md, int Nd, int Kd, int mode) {
        int groups = (Md / 16) * (Nd / 64);
        wmma_gemm_tn<4><<<(groups + 3) / 4, 128, 0, stream>>>(A, Bt, bias, of, oh, Md, Nd, Kd, mode);
    };
    auto gemm2 = [&](const _Float16* A, const _Float16* Bt, const float* bias,
                     float* of, _Float16* oh, int Md, int Nd, int Kd, int mode) {
        int groups = (Md / 16) * (Nd / 32);
        wmma_gemm_tn<2><<<(groups + 3) / 4, 128, 0, stream>>>(A, Bt, bias, of, oh, Md, Nd, Kd, mode);
    };

    // ---- embedding ----
    embed_kernel<<<(BN * HQ + 255) / 256, 256, 0, stream>>>(X, t, node_emb, time_emb, x);

    // ---- transformer layers ----
    for (int l = 0; l < 12; ++l) {
        const float* g1  = ln1_g + l * HQ;
        const float* b1  = ln1_b + l * HQ;
        const float* g2  = ln2_g + l * HQ;
        const float* b2l = ln2_b + l * HQ;
        const float* ee  = edge_emb + l * EVQ * NHQ;

        ln_f16_kernel<<<BN, 256, 0, stream>>>(x, g1, b1, h16);

        convT(Wq + (size_t)l * HQ * HQ, wt, HQ, HQ);
        gemm4(h16, wt, bq + l * HQ, nullptr, q16, BN, HQ, HQ, 0);
        convT(Wk + (size_t)l * HQ * HQ, wt, HQ, HQ);
        gemm4(h16, wt, bk + l * HQ, nullptr, k16, BN, HQ, HQ, 0);
        convT(Wv + (size_t)l * HQ * HQ, wt, HQ, HQ);
        gemm4(h16, wt, bv + l * HQ, nullptr, v16, BN, HQ, HQ, 0);

        attn_kernel<<<BQ * NHQ * 4, 128, 0, stream>>>(q16, k16, v16, E, Mm, ee, ao16);

        convT(Wo + (size_t)l * HQ * HQ, wt, HQ, HQ);
        gemm4(ao16, wt, bo + l * HQ, x, nullptr, BN, HQ, HQ, 2);   // x += attn_out

        ln_f16_kernel<<<BN, 256, 0, stream>>>(x, g2, b2l, h16);

        convT(Wf1 + (size_t)l * HQ * FFQ, wt, HQ, FFQ);
        gemm4(h16, wt, bf1 + l * FFQ, nullptr, ff16, BN, FFQ, HQ, 1);  // gelu, f16
        convT(Wf2 + (size_t)l * FFQ * HQ, wt, FFQ, HQ);
        gemm4(ff16, wt, bf2 + l * HQ, x, nullptr, BN, HQ, FFQ, 2);     // x += ffn
    }

    // ---- final LN ----
    ln_f16_kernel<<<BN, 256, 0, stream>>>(x, lnf_g, lnf_b, xln16);

    // ---- node head: logits -> d_out[0 .. 32768) ----
    convT(node_W, wt, HQ, 32);
    gemm2(xln16, wt, nullptr, out, nullptr, BN, 32, HQ, 3);

    // ---- edge head ----
    convT(eh_W1, wt, HQ, HQ);                        // W1[:H]
    gemm4(xln16, wt, eh_b1, ai, nullptr, BN, HQ, HQ, 3);
    convT(eh_W1 + (size_t)HQ * HQ, wt, HQ, HQ);      // W1[H:]
    gemm4(xln16, wt, nullptr, aj, nullptr, BN, HQ, HQ, 3);

    edge_head_kernel<<<BQ * NQ, 128, 0, stream>>>(ai, aj, eh_W2, eh_b2, elraw);
    symm_kernel<<<(BQ * NQ * NQ * EVQ + 255) / 256, 256, 0, stream>>>(elraw, out + BN * 32);
}
